// NoCSkip_20547123544826
// MI455X (gfx1250) — matrix operand order repaired and checked by reading, not compiled, not run on hardware
//
#include <hip/hip_runtime.h>

typedef __attribute__((ext_vector_type(16))) _Float16 v16h;
typedef __attribute__((ext_vector_type(8)))  float    v8f;

// ---------------- workspace layout (float offsets) ----------------
static constexpr int P_KEEP    = 0;          // 20001 (pad 20032)
static constexpr int TGT_PART  = 20032;      // 512
static constexpr int TGT2B     = 20544;      // 512
static constexpr int TIME_ENC  = 21056;      // 64*50*512
static constexpr int TF2       = 1659456;    // 64*50*64
static constexpr int SELECTED  = 1864256;    // 64*50*512
static constexpr int XG        = 3502656;    // 64*50*256
static constexpr int FOLLOWING = 4321856;    // 50*64*64
static constexpr int MEM       = 4526656;    // 50*64*512
static constexpr int HIDDENS   = 6165056;    // 64*50*512
static constexpr int HX        = 7803456;    // 64*512
static constexpr int VIN       = 7836224;    // 64*1600
static constexpr int TMP1      = 7938624;    // 64*512
static constexpr int V0BUF     = 7971392;    // 64
static constexpr int ZBUF      = 7971456;    // 64*512
static constexpr int XG3       = 8004224;    // 64*1536
static constexpr int HG3       = 8102528;    // 64*1536
static constexpr int STEPH     = 8200832;    // 64*512
static constexpr int SQ        = 8233600;    // 64*512
static constexpr int H2        = 8266368;    // 64*512
static constexpr int FF1O      = 8299136;    // 64*2048
static constexpr int HFF       = 8430208;    // 64*512
static constexpr int MPRE      = 8462976;    // 64*512
static constexpr int GTMP      = 8495744;    // 64*50*512
// f16 packed regions (sizes in floats = halves/2)
static constexpr int APACK = 10134144;  // 65536  (131072 halves scratch)
static constexpr int SELP  = 10199680;  // 819200
static constexpr int CS1P  = 11018880;  // 131072
static constexpr int LWIP  = 11149952;  // 65536
static constexpr int VS1P  = 11215488;  // 409600
static constexpr int FCP   = 11625088;  // 131072
static constexpr int GWIP  = 11756160;  // 393216
static constexpr int GWHP  = 12149376;  // 393216
static constexpr int SLP   = 12542592;  // 131072
static constexpr int FF1P  = 12673664;  // 524288
static constexpr int FF2P  = 13197952;  // 524288
static constexpr int MLP   = 13722240;  // 131072
static constexpr int GA1P  = 13853312;  // 278528

__device__ __forceinline__ float sigf(float x) { return 1.f / (1.f + __expf(-x)); }

// ---- WMMA tile from pre-packed f16 operands.
// Packed layout for both A ([mtiles][ktiles][32 lanes][16]) and B ([tilesN][ktiles][32][16]).
// unroll 4: keeps 4 operand pairs in flight (~72 VGPRs live) without spilling.
__device__ __forceinline__ v8f wmma_tile_pk(const _Float16* __restrict__ Ap,
                                            const _Float16* __restrict__ Bp,
                                            int mt, int tn, int ktiles) {
  const int l = threadIdx.x & 31;
  v8f acc = {0.f, 0.f, 0.f, 0.f, 0.f, 0.f, 0.f, 0.f};
  const _Float16* ap = Ap + ((size_t)(mt * ktiles) * 32 + l) * 16;
  const _Float16* bp = Bp + ((size_t)(tn * ktiles) * 32 + l) * 16;
#pragma unroll 4
  for (int kt = 0; kt < ktiles; ++kt) {
    v16h a = *(const v16h*)ap; ap += 512;   // 32 lanes * 16 halves
    v16h b = *(const v16h*)bp; bp += 512;
    acc = __builtin_amdgcn_wmma_f32_16x16x32_f16(false, a, false, b, (short)0, acc, false, false);
  }
  return acc;
}

// act: 0=none 1=relu 2=tanh
__device__ __forceinline__ void store_tile(float* __restrict__ C, int ldc, int m0, int n0,
                                           int M, v8f acc, const float* __restrict__ bias, int act) {
  const int l = threadIdx.x & 31;
  const int n = n0 + (l & 15);
  const int mb = m0 + ((l >> 4) << 3);
#pragma unroll
  for (int r = 0; r < 8; ++r) {
    const int m = mb + r;
    if (m < M) {
      float v = acc[r];
      if (bias) v += bias[n];
      if (act == 1) v = fmaxf(v, 0.f);
      else if (act == 2) v = tanhf(v);
      C[m * ldc + n] = v;
    }
  }
}

// ---------------- weight pre-pack: B(KxN f32,row-major) -> swizzled f16 ----------------
__global__ void k_pack_b(const float* __restrict__ B, int ldb, int K, int N,
                         _Float16* __restrict__ dst) {
  const int ktiles = K >> 5;
  const int total = (N >> 4) * ktiles * 512;
  for (int idx = blockIdx.x * blockDim.x + threadIdx.x; idx < total;
       idx += gridDim.x * blockDim.x) {
    const int tn = idx / (ktiles * 512);
    const int rem = idx - tn * (ktiles * 512);
    const int kt = rem >> 9;
    const int rr = rem & 511;
    const int ll = rr >> 4, j = rr & 15;
    const int k = (kt << 5) + ((j >> 3) << 4) + ((ll >> 4) << 3) + (j & 7);
    const int n = (tn << 4) + (ll & 15);
    dst[idx] = (_Float16)B[k * ldb + n];
  }
}

// ---------------- activation pre-pack: A(MxK f32) -> swizzled f16 (grid version) ----------------
__global__ void k_pack_a(const float* __restrict__ A, int lda, int mtiles, int K,
                         _Float16* __restrict__ dst) {
  const int ktiles = K >> 5;
  const int total = mtiles * ktiles * 512;
  for (int idx = blockIdx.x * blockDim.x + threadIdx.x; idx < total;
       idx += gridDim.x * blockDim.x) {
    const int mt = idx / (ktiles * 512);
    const int rem = idx - mt * (ktiles * 512);
    const int kt = rem >> 9;
    const int rr = rem & 511;
    const int ll = rr >> 4, j = rr & 15;
    const int m = (mt << 4) + (ll & 15);
    const int k = (kt << 5) + ((j >> 3) << 4) + ((ll >> 4) << 3) + (j & 7);
    dst[idx] = (_Float16)A[m * lda + k];
  }
}

// workgroup-internal A pack (1024 threads), same layout
__device__ __forceinline__ void pack_A_wg(const float* __restrict__ A, int lda, int mtiles,
                                          int K, _Float16* __restrict__ dst) {
  const int ktiles = K >> 5;
  const int total = mtiles * ktiles * 512;
  for (int idx = threadIdx.x; idx < total; idx += 1024) {
    const int mt = idx / (ktiles * 512);
    const int rem = idx - mt * (ktiles * 512);
    const int kt = rem >> 9;
    const int rr = rem & 511;
    const int ll = rr >> 4, j = rr & 15;
    const int m = (mt << 4) + (ll & 15);
    const int k = (kt << 5) + ((j >> 3) << 4) + ((ll >> 4) << 3) + (j & 7);
    dst[idx] = (_Float16)A[m * lda + k];
  }
}

// ---------------- K0: constant vectors ----------------
__global__ void k_consts(const float* tgt, const float* cs1_w, const float* tl2_w,
                         const float* tl2_b, float* ws) {
  const int n = threadIdx.x; // 512
  float a = 0.f, b2 = tl2_b[n];
  for (int k = 0; k < 512; ++k) {
    const float tk = tgt[k];
    a  += tk * cs1_w[(512 + k) * 512 + n];
    b2 += tk * tl2_w[k * 512 + n];
  }
  ws[TGT_PART + n] = a;
  ws[TGT2B + n]    = b2;
}

// ---------------- K1: vocabulary code-selection, LDS-staged A + packed B ----------------
__global__ void k_code_select(const float* __restrict__ emb, const _Float16* __restrict__ cs1p,
                              const float* cs1_b, const float* cs2_w, const float* cs2_b,
                              const float* g_code, float* ws) {
  __shared__ alignas(32) _Float16 lds_a[16 * 512];   // one 16-row strip, swizzled f16
  __shared__ float pl[16][2];
  const int w = threadIdx.x >> 5;   // 8 waves
  const int l = threadIdx.x & 31;
  if (threadIdx.x < 32) pl[threadIdx.x >> 1][threadIdx.x & 1] = 0.f;
  const int m0 = blockIdx.x << 4;
  // stage + convert + swizzle A strip cooperatively (layout [kt][lane][16])
  for (int idx = threadIdx.x; idx < 8192; idx += 256) {
    const int kt = idx >> 9;
    const int rr = idx & 511;
    const int ll = rr >> 4, j = rr & 15;
    int m = m0 + (ll & 15); if (m > 20000) m = 20000;
    const int k = (kt << 5) + ((j >> 3) << 4) + ((ll >> 4) << 3) + (j & 7);
    lds_a[idx] = (_Float16)emb[m * 512 + k];
  }
  __syncthreads();
  for (int tn = 0; tn < 4; ++tn) {
    const int tnG = (w << 2) + tn;          // 0..31
    v8f acc = {0.f, 0.f, 0.f, 0.f, 0.f, 0.f, 0.f, 0.f};
    const _Float16* ap = lds_a + l * 16;
    const _Float16* bp = cs1p + ((size_t)(tnG * 16) * 32 + l) * 16;
#pragma unroll 4
    for (int kt = 0; kt < 16; ++kt) {
      v16h a = *(const v16h*)ap; ap += 512;
      v16h b = *(const v16h*)bp; bp += 512;
      acc = __builtin_amdgcn_wmma_f32_16x16x32_f16(false, a, false, b, (short)0, acc, false, false);
    }
    const int n = (tnG << 4) + (l & 15);
    const float c0 = cs2_w[n * 2 + 0], c1 = cs2_w[n * 2 + 1];
    const float add = cs1_b[n] + ws[TGT_PART + n];
#pragma unroll
    for (int r = 0; r < 8; ++r) {
      const float h = tanhf(acc[r] + add);
      const int mrow = ((l >> 4) << 3) + r;
      atomicAdd(&pl[mrow][0], h * c0);
      atomicAdd(&pl[mrow][1], h * c1);
    }
  }
  __syncthreads();
  if (threadIdx.x < 16) {
    const int m = m0 + threadIdx.x;
    if (m < 20001) {
      const float p0 = pl[threadIdx.x][0] + cs2_b[0] + g_code[m * 2 + 0];
      const float p1 = pl[threadIdx.x][1] + cs2_b[1] + g_code[m * 2 + 1];
      ws[P_KEEP + m] = (p0 >= p1) ? 1.f : 0.f;
    }
  }
}

// ---------------- K2: time encodings ----------------
__global__ void k_time(const float* stime, const float* t1_w, const float* t1_b,
                       const float* t2_w, const float* t2_b,
                       const float* tu_w, const float* tu_b, float* ws) {
  __shared__ float tf[64];
  const int bs = blockIdx.x;
  const float t = stime[bs] * (1.f / 180.f);
  if (threadIdx.x < 64) {
    const int j = threadIdx.x;
    const float u1 = t * t1_w[j] + t1_b[j];
    tf[j] = 1.f - tanhf(u1 * u1);
    const float u2 = t * t2_w[j] + t2_b[j];
    ws[TF2 + bs * 64 + j] = 1.f - tanhf(u2 * u2);
  }
  __syncthreads();
  const int d = threadIdx.x;
  float acc = tu_b[d];
  for (int j = 0; j < 64; ++j) acc += tf[j] * tu_w[j * 512 + d];
  ws[TIME_ENC + bs * 512 + d] = acc;
}

// ---------------- K3: selected = sum_c mask*relu(emb) + time_enc ----------------
__global__ void k_selected(const int* seqs, const float* emb, float* ws) {
  const int bs = blockIdx.x;
  const int d = threadIdx.x;
  float acc = ws[TIME_ENC + bs * 512 + d];
  for (int c = 0; c < 20; ++c) {
    const int v = seqs[bs * 20 + c];
    const float m = ws[P_KEEP + v];
    acc += m * fmaxf(emb[v * 512 + d], 0.f);
  }
  ws[SELECTED + bs * 512 + d] = acc;
}

// ---------------- generic GEMM over packed f16 operands ----------------
__global__ void gemm_pk(const _Float16* __restrict__ Ap, const _Float16* __restrict__ Bp,
                        const float* bias, float* C, int ldc, int M, int N, int K, int act) {
  const int ktiles = K >> 5;
  const int tilesN = N >> 4;
  const int tilesM = (M + 15) >> 4;
  const int total = tilesM * tilesN;
  const int gw = blockIdx.x * (blockDim.x >> 5) + (threadIdx.x >> 5);
  const int nw = gridDim.x * (blockDim.x >> 5);
  for (int t = gw; t < total; t += nw) {
    const int tm = t / tilesN, tn = t - tm * tilesN;
    v8f acc = wmma_tile_pk(Ap, Bp, tm, tn, ktiles);
    store_tile(C, ldc, tm << 4, tn << 4, M, acc, bias, act);
  }
}

// ---------------- K4b: backward LSTM (recurrent part only, H=64) ----------------
__global__ void k_lstm(const float* wh, float* ws) {
  __shared__ float h[4096], c[4096];
  const int t = threadIdx.x; // 256
  for (int p = t; p < 4096; p += 256) { h[p] = 0.f; c[p] = 0.f; }
  __syncthreads();
  for (int ii = 0; ii < 50; ++ii) {
    const int s = 49 - ii;
    float hn[16], cn[16];
#pragma unroll 1
    for (int q = 0; q < 16; ++q) {
      const int p = t * 16 + q;
      const int b = p >> 6, j = p & 63;
      float g0 = ws[XG + (b * 50 + s) * 256 + j];
      float g1 = ws[XG + (b * 50 + s) * 256 + 64 + j];
      float g2 = ws[XG + (b * 50 + s) * 256 + 128 + j];
      float g3 = ws[XG + (b * 50 + s) * 256 + 192 + j];
      for (int k = 0; k < 64; ++k) {
        const float hk = h[b * 64 + k];
        const float* wr = wh + k * 256 + j;
        g0 += hk * wr[0]; g1 += hk * wr[64]; g2 += hk * wr[128]; g3 += hk * wr[192];
      }
      const float ci = sigf(g1) * c[p] + sigf(g0) * tanhf(g2);
      cn[q] = ci; hn[q] = sigf(g3) * tanhf(ci);
    }
    __syncthreads();
    for (int q = 0; q < 16; ++q) {
      const int p = t * 16 + q;
      const int b = p >> 6, j = p & 63;
      h[p] = hn[q]; c[p] = cn[q];
      ws[FOLLOWING + (s * 64 + b) * 64 + j] = hn[q];
    }
    __syncthreads();
  }
}

#define FSYNC() do { __threadfence_block(); __syncthreads(); } while (0)

__device__ __forceinline__ void ln_rows(float* __restrict__ X, const float* __restrict__ g,
                                        const float* __restrict__ bb,
                                        float* sh_m, float* sh_v) {
  const int tid = threadIdx.x;
  FSYNC();
  if (tid < 64) {
    const float* row = X + tid * 512;
    float s = 0.f;
    for (int k = 0; k < 512; ++k) s += row[k];
    const float m = s * (1.f / 512.f);
    float v = 0.f;
    for (int k = 0; k < 512; ++k) { const float d = row[k] - m; v += d * d; }
    sh_m[tid] = m;
    sh_v[tid] = 1.f / sqrtf(v * (1.f / 512.f) + 1e-5f);
  }
  __syncthreads();
  for (int p = tid; p < 64 * 512; p += 1024) {
    const int b = p >> 9, d = p & 511;
    X[p] = (X[p] - sh_m[b]) * sh_v[b] * g[d] + bb[d];
  }
  FSYNC();
}

// ---------------- K5: sequential 50-step scan, one persistent 32-wave workgroup ----------------
__global__ void __launch_bounds__(1024) k_scan(
    const float* vs1_b, const float* vs2_w, const float* vs2_b,
    const float* g_skip, const float* fc_b,
    const float* sl_b, const float* sl_g, const float* sl_bn,
    const float* ln2_g, const float* ln2_b,
    const float* ff1_b, const float* ff2_b,
    const float* ln_g, const float* ln_b,
    const float* ml_b, const float* ml_g, const float* ml_bn,
    _Float16* apk,
    const _Float16* vs1p, const _Float16* fcp, const _Float16* gwip, const _Float16* gwhp,
    const _Float16* slp, const _Float16* ff1p, const _Float16* ff2p, const _Float16* mlp,
    float* ws) {
  const int tid = threadIdx.x;
  const int w = tid >> 5;
  const int l = tid & 31;
  __shared__ float sh_e[64 * 50];
  __shared__ float sh_m[64], sh_v[64];

  for (int p = tid; p < 50 * 64 * 512; p += 1024) ws[MEM + p] = 0.f;
  for (int p = tid; p < 64 * 512; p += 1024) ws[HX + p] = 0.f;
  FSYNC();

  for (int i = 0; i < 50; ++i) {
    // A: build vin = [hx | sel_i | following_i | tgt2b]  (64 x 1600)
    for (int p = tid; p < 64 * 1600; p += 1024) {
      const int b = p / 1600, k = p - b * 1600;
      float v;
      if (k < 512)       v = ws[HX + b * 512 + k];
      else if (k < 1024) v = ws[SELECTED + (b * 50 + i) * 512 + (k - 512)];
      else if (k < 1088) v = ws[FOLLOWING + (i * 64 + b) * 64 + (k - 1024)];
      else               v = ws[TGT2B + (k - 1088)];
      ws[VIN + p] = v;
    }
    FSYNC();
    pack_A_wg(ws + VIN, 1600, 4, 1600, apk);
    FSYNC();
    // B: tmp1 = tanh(vin @ vs1_w + vs1_b)
    for (int t = w; t < 128; t += 32) {
      const int tm = t >> 5, tn = t & 31;
      v8f acc = wmma_tile_pk(apk, vs1p, tm, tn, 50);
      store_tile(ws + TMP1, 512, tm << 4, tn << 4, 64, acc, vs1_b, 2);
    }
    FSYNC();
    // C: hard skip gate  v0 = [argmax(vlog + g) == 0]
    if (tid < 64) {
      const int b = tid;
      float p0 = vs2_b[0], p1 = vs2_b[1];
      for (int k = 0; k < 512; ++k) {
        const float h = ws[TMP1 + b * 512 + k];
        p0 += h * vs2_w[k * 2 + 0];
        p1 += h * vs2_w[k * 2 + 1];
      }
      p0 += g_skip[(i * 64 + b) * 2 + 0];
      p1 += g_skip[(i * 64 + b) * 2 + 1];
      ws[V0BUF + b] = (p0 >= p1) ? 1.f : 0.f;
    }
    pack_A_wg(ws + SELECTED + i * 512, 50 * 512, 4, 512, apk);
    FSYNC();
    // D: z = sel_i @ fc_w + fc_b + time_enc_i
    for (int t = w; t < 128; t += 32) {
      const int tm = t >> 5, tn = t & 31;
      v8f acc = wmma_tile_pk(apk, fcp, tm, tn, 16);
      const int n = (tn << 4) + (l & 15);
      const int mb = (tm << 4) + ((l >> 4) << 3);
#pragma unroll
      for (int r = 0; r < 8; ++r) {
        const int m = mb + r;
        ws[ZBUF + m * 512 + n] = acc[r] + fc_b[n] + ws[TIME_ENC + (m * 50 + i) * 512 + n];
      }
    }
    FSYNC();
    pack_A_wg(ws + ZBUF, 512, 4, 512, apk);
    pack_A_wg(ws + HX, 512, 4, 512, apk + 32768);
    FSYNC();
    // E: xg3 = z @ gru_wi ; hg3 = hx @ gru_wh   (64x1536 each)
    for (int t = w; t < 768; t += 32) {
      const int which = (t >= 384);
      const int tt = t - which * 384;
      const int tm = tt / 96, tn = tt - tm * 96;
      v8f acc = wmma_tile_pk(apk + which * 32768, which ? gwhp : gwip, tm, tn, 16);
      store_tile(which ? (ws + HG3) : (ws + XG3), 1536, tm << 4, tn << 4, 64, acc, nullptr, 0);
    }
    FSYNC();
    // F: GRU elementwise + hard skip mix
    for (int p = tid; p < 64 * 512; p += 1024) {
      const int b = p >> 9, d = p & 511;
      const float r = sigf(ws[XG3 + b * 1536 + d] + ws[HG3 + b * 1536 + d]);
      const float u = sigf(ws[XG3 + b * 1536 + 512 + d] + ws[HG3 + b * 1536 + 512 + d]);
      const float n = tanhf(ws[XG3 + b * 1536 + 1024 + d] + r * ws[HG3 + b * 1536 + 1024 + d]);
      const float hx = ws[HX + p];
      const float gruh = (1.f - u) * n + u * hx;
      const float v0 = ws[V0BUF + b];
      ws[STEPH + p] = hx * (1.f - v0) + gruh * v0;
    }
    FSYNC();
    pack_A_wg(ws + STEPH, 512, 4, 512, apk);
    FSYNC();
    // G: sq = LN(relu(steph @ sl_w + sl_b))
    for (int t = w; t < 128; t += 32) {
      const int tm = t >> 5, tn = t & 31;
      v8f acc = wmma_tile_pk(apk, slp, tm, tn, 16);
      store_tile(ws + SQ, 512, tm << 4, tn << 4, 64, acc, sl_b, 1);
    }
    ln_rows(ws + SQ, sl_g, sl_bn, sh_m, sh_v);
    // H: masked attention over mem rows < i
    for (int p = tid; p < 64 * 50; p += 1024) {
      const int b = p / 50, s = p - b * 50;
      float e = -1e9f;
      if (s < i) {
        float acc = 0.f;
        for (int k = 0; k < 512; ++k)
          acc += ws[SQ + b * 512 + k] * ws[MEM + (s * 64 + b) * 512 + k];
        e = acc * (1.f / 22.62741699796952f);
      }
      sh_e[b * 50 + s] = e;
    }
    __syncthreads();
    if (tid < 64) {
      float mx = -3.4e38f;
      for (int s = 0; s < 50; ++s) mx = fmaxf(mx, sh_e[tid * 50 + s]);
      float sum = 0.f;
      for (int s = 0; s < 50; ++s) { const float e = __expf(sh_e[tid * 50 + s] - mx); sh_e[tid * 50 + s] = e; sum += e; }
      const float inv = 1.f / sum;
      for (int s = 0; s < 50; ++s) sh_e[tid * 50 + s] *= inv;
    }
    __syncthreads();
    for (int p = tid; p < 64 * 512; p += 1024) {
      const int b = p >> 9, d = p & 511;
      float acc = ws[SQ + p];
      for (int s = 0; s < 50; ++s) acc += sh_e[b * 50 + s] * ws[MEM + (s * 64 + b) * 512 + d];
      ws[H2 + p] = acc;
    }
    ln_rows(ws + H2, ln2_g, ln2_b, sh_m, sh_v);
    pack_A_wg(ws + H2, 512, 4, 512, apk);
    FSYNC();
    // I: ff1o = relu(h2 @ ff1 + b)   (64x2048)
    for (int t = w; t < 512; t += 32) {
      const int tm = t >> 7, tn = t & 127;
      v8f acc = wmma_tile_pk(apk, ff1p, tm, tn, 16);
      store_tile(ws + FF1O, 2048, tm << 4, tn << 4, 64, acc, ff1_b, 1);
    }
    FSYNC();
    pack_A_wg(ws + FF1O, 2048, 4, 2048, apk);
    FSYNC();
    // J: hff_pre = ff1o @ ff2 + b + h2
    for (int t = w; t < 128; t += 32) {
      const int tm = t >> 5, tn = t & 31;
      v8f acc = wmma_tile_pk(apk, ff2p, tm, tn, 64);
      const int n = (tn << 4) + (l & 15);
      const int mb = (tm << 4) + ((l >> 4) << 3);
#pragma unroll
      for (int r = 0; r < 8; ++r) {
        const int m = mb + r;
        ws[HFF + m * 512 + n] = acc[r] + ff2_b[n] + ws[H2 + m * 512 + n];
      }
    }
    ln_rows(ws + HFF, ln_g, ln_b, sh_m, sh_v);
    // K: hx_new = (i==0) ? steph : hff ; record hiddens
    for (int p = tid; p < 64 * 512; p += 1024) {
      const float hn = (i == 0) ? ws[STEPH + p] : ws[HFF + p];
      ws[HX + p] = hn;
      const int b = p >> 9, d = p & 511;
      ws[HIDDENS + (b * 50 + i) * 512 + d] = hn;
    }
    FSYNC();
    pack_A_wg(ws + HX, 512, 4, 512, apk);
    FSYNC();
    // L: mem[i] = LN(relu(hx_new @ ml + b))
    for (int t = w; t < 128; t += 32) {
      const int tm = t >> 5, tn = t & 31;
      v8f acc = wmma_tile_pk(apk, mlp, tm, tn, 16);
      store_tile(ws + MPRE, 512, tm << 4, tn << 4, 64, acc, ml_b, 1);
    }
    ln_rows(ws + MPRE, ml_g, ml_bn, sh_m, sh_v);
    for (int p = tid; p < 64 * 512; p += 1024) {
      const int b = p >> 9, d = p & 511;
      ws[MEM + (i * 64 + b) * 512 + d] = ws[MPRE + p];
    }
    FSYNC();
  }
}

// ---------------- K6: gtmp = tanh(gin @ ga1 + b), gathered A + packed B ----------------
__global__ void k_ga1(const _Float16* __restrict__ ga1p, const float* ga1_b,
                      const float* tgt, float* ws) {
  const int gw = blockIdx.x * (blockDim.x >> 5) + (threadIdx.x >> 5);
  const int nw = gridDim.x * (blockDim.x >> 5);
  const int l = threadIdx.x & 31, lm = l & 15, lh = l >> 4;
  for (int t = gw; t < 6400; t += nw) {   // 200 x 32 tiles, K=1088 (34 ktiles)
    const int tm = t >> 5, tn = t & 31;
    const int mA = (tm << 4) + lm;
    v8f acc = {0.f, 0.f, 0.f, 0.f, 0.f, 0.f, 0.f, 0.f};
    const _Float16* bp = ga1p + ((size_t)(tn * 34) * 32 + l) * 16;
#pragma unroll 2
    for (int kt = 0; kt < 34; ++kt) {
      v16h a, b;
      b = *(const v16h*)bp; bp += 512;
#pragma unroll
      for (int j = 0; j < 16; ++j) {
        const int ka = (kt << 5) + ((j >> 3) << 4) + (lh << 3) + (j & 7);
        float av;
        if (ka < 512)      av = ws[HIDDENS + mA * 512 + ka];
        else if (ka < 576) av = ws[TF2 + mA * 64 + (ka - 512)];
        else               av = tgt[ka - 576];
        a[j] = (_Float16)av;
      }
      acc = __builtin_amdgcn_wmma_f32_16x16x32_f16(false, a, false, b, (short)0, acc, false, false);
    }
    store_tile(ws + GTMP, 512, tm << 4, tn << 4, 3200, acc, ga1_b, 2);
  }
}

// ---------------- K7: gate softmax + pool + output head ----------------
__global__ void k_final(const float* ga2_w, const float* ga2_b, const int* lengths,
                        const float* out_w, const float* out_b, float* out, float* ws) {
  __shared__ float ge[50], of[512];
  const int b = blockIdx.x, t = threadIdx.x;
  if (t < 50) {
    float acc = ga2_b[0];
    for (int k = 0; k < 512; ++k) acc += ws[GTMP + (b * 50 + t) * 512 + k] * ga2_w[k];
    ge[t] = (t >= lengths[b]) ? -1e9f : acc;
  }
  __syncthreads();
  if (t == 0) {
    float mx = -3.4e38f;
    for (int s = 0; s < 50; ++s) mx = fmaxf(mx, ge[s]);
    float sum = 0.f;
    for (int s = 0; s < 50; ++s) { const float e = __expf(ge[s] - mx); ge[s] = e; sum += e; }
    const float inv = 1.f / sum;
    for (int s = 0; s < 50; ++s) ge[s] *= inv;
  }
  __syncthreads();
  {
    const int d = t;
    float acc = 0.f;
    for (int s = 0; s < 50; ++s) acc += ge[s] * ws[HIDDENS + (b * 50 + s) * 512 + d];
    of[d] = acc;
  }
  __syncthreads();
  if (t < 2) {
    float acc = out_b[t];
    for (int k = 0; k < 512; ++k) acc += of[k] * out_w[k * 2 + t];
    out[b * 2 + t] = acc;
  }
}

extern "C" void kernel_launch(void* const* d_in, const int* in_sizes, int n_in,
                              void* d_out, int out_size, void* d_ws, size_t ws_size,
                              hipStream_t stream) {
  (void)in_sizes; (void)n_in; (void)out_size; (void)ws_size;
  const int*   input_seqs = (const int*)d_in[0];
  const int*   lengths    = (const int*)d_in[1];
  const float* stime      = (const float*)d_in[2];
  const float* emb    = (const float*)d_in[3];
  const float* tgt    = (const float*)d_in[4];
  const float* tl2_w  = (const float*)d_in[5];
  const float* tl2_b  = (const float*)d_in[6];
  const float* cs1_w  = (const float*)d_in[7];
  const float* cs1_b  = (const float*)d_in[8];
  const float* cs2_w  = (const float*)d_in[9];
  const float* cs2_b  = (const float*)d_in[10];
  const float* lstm_wi = (const float*)d_in[11];
  const float* lstm_wh = (const float*)d_in[12];
  const float* vs1_w  = (const float*)d_in[13];
  const float* vs1_b  = (const float*)d_in[14];
  const float* vs2_w  = (const float*)d_in[15];
  const float* vs2_b  = (const float*)d_in[16];
  const float* out_w  = (const float*)d_in[17];
  const float* out_b  = (const float*)d_in[18];
  const float* ln_g   = (const float*)d_in[19];
  const float* ln_b   = (const float*)d_in[20];
  const float* ln2_g  = (const float*)d_in[21];
  const float* ln2_b  = (const float*)d_in[22];
  const float* fc_w   = (const float*)d_in[23];
  const float* fc_b   = (const float*)d_in[24];
  const float* ff1_w  = (const float*)d_in[25];
  const float* ff1_b  = (const float*)d_in[26];
  const float* ff2_w  = (const float*)d_in[27];
  const float* ff2_b  = (const float*)d_in[28];
  const float* gru_wi = (const float*)d_in[29];
  const float* gru_wh = (const float*)d_in[30];
  const float* t1_w   = (const float*)d_in[31];
  const float* t1_b   = (const float*)d_in[32];
  const float* t2_w   = (const float*)d_in[33];
  const float* t2_b   = (const float*)d_in[34];
  const float* tu_w   = (const float*)d_in[35];
  const float* tu_b   = (const float*)d_in[36];
  const float* ml_w   = (const float*)d_in[37];
  const float* ml_b   = (const float*)d_in[38];
  const float* ml_g   = (const float*)d_in[39];
  const float* ml_bn  = (const float*)d_in[40];
  const float* sl_w   = (const float*)d_in[41];
  const float* sl_b   = (const float*)d_in[42];
  const float* sl_g   = (const float*)d_in[43];
  const float* sl_bn  = (const float*)d_in[44];
  const float* ga1_w  = (const float*)d_in[45];
  const float* ga1_b  = (const float*)d_in[46];
  const float* ga2_w  = (const float*)d_in[47];
  const float* ga2_b  = (const float*)d_in[48];
  const float* g_code = (const float*)d_in[49];
  const float* g_skip = (const float*)d_in[50];
  float* out = (float*)d_out;
  float* ws  = (float*)d_ws;

  _Float16* apk  = (_Float16*)(ws + APACK);
  _Float16* selp = (_Float16*)(ws + SELP);
  _Float16* cs1p = (_Float16*)(ws + CS1P);
  _Float16* lwip = (_Float16*)(ws + LWIP);
  _Float16* vs1p = (_Float16*)(ws + VS1P);
  _Float16* fcp  = (_Float16*)(ws + FCP);
  _Float16* gwip = (_Float16*)(ws + GWIP);
  _Float16* gwhp = (_Float16*)(ws + GWHP);
  _Float16* slp  = (_Float16*)(ws + SLP);
  _Float16* ff1p = (_Float16*)(ws + FF1P);
  _Float16* ff2p = (_Float16*)(ws + FF2P);
  _Float16* mlp  = (_Float16*)(ws + MLP);
  _Float16* ga1p = (_Float16*)(ws + GA1P);

  // weight pre-pack (f32 -> swizzled f16)
  k_pack_b<<<128, 256, 0, stream>>>(cs1_w, 512, 512, 512, cs1p);
  k_pack_b<<<128, 256, 0, stream>>>(lstm_wi, 256, 512, 256, lwip);
  k_pack_b<<<128, 256, 0, stream>>>(vs1_w, 512, 1600, 512, vs1p);
  k_pack_b<<<128, 256, 0, stream>>>(fc_w, 512, 512, 512, fcp);
  k_pack_b<<<128, 256, 0, stream>>>(gru_wi, 1536, 512, 1536, gwip);
  k_pack_b<<<128, 256, 0, stream>>>(gru_wh, 1536, 512, 1536, gwhp);
  k_pack_b<<<128, 256, 0, stream>>>(sl_w, 512, 512, 512, slp);
  k_pack_b<<<128, 256, 0, stream>>>(ff1_w, 2048, 512, 2048, ff1p);
  k_pack_b<<<128, 256, 0, stream>>>(ff2_w, 512, 2048, 512, ff2p);
  k_pack_b<<<128, 256, 0, stream>>>(ml_w, 512, 512, 512, mlp);
  k_pack_b<<<128, 256, 0, stream>>>(ga1_w, 512, 1088, 512, ga1p);

  k_consts<<<1, 512, 0, stream>>>(tgt, cs1_w, tl2_w, tl2_b, ws);
  k_code_select<<<1251, 256, 0, stream>>>(emb, cs1p, cs1_b, cs2_w, cs2_b, g_code, ws);
  k_time<<<3200, 512, 0, stream>>>(stime, t1_w, t1_b, t2_w, t2_b, tu_w, tu_b, ws);
  k_selected<<<3200, 512, 0, stream>>>(input_seqs, emb, ws);
  // pack selected, then XG = selected @ lstm_wi
  k_pack_a<<<400, 256, 0, stream>>>(ws + SELECTED, 512, 200, 512, selp);
  gemm_pk<<<400, 256, 0, stream>>>(selp, lwip, nullptr, ws + XG, 256, 3200, 256, 512, 0);
  k_lstm<<<1, 256, 0, stream>>>(lstm_wh, ws);
  k_scan<<<1, 1024, 0, stream>>>(vs1_b, vs2_w, vs2_b, g_skip, fc_b,
                                 sl_b, sl_g, sl_bn, ln2_g, ln2_b,
                                 ff1_b, ff2_b, ln_g, ln_b,
                                 ml_b, ml_g, ml_bn,
                                 apk, vs1p, fcp, gwip, gwhp, slp, ff1p, ff2p, mlp, ws);
  k_ga1<<<800, 256, 0, stream>>>(ga1p, ga1_b, tgt, ws);
  k_final<<<64, 512, 0, stream>>>(ga2_w, ga2_b, lengths, out_w, out_b, out, ws);
}
